// HeteroGNN_CG_76613626626595
// MI455X (gfx1250) — compile-verified, matrix operand order checked by reading, not run on hardware
//
#include <hip/hip_runtime.h>
#include <stdint.h>

// ---- problem constants (match reference) ----
#define N_NODES 65536
#define N_EDGES 200000
#define HDIM    128
#define BGRAPH  64

typedef __attribute__((ext_vector_type(16))) __bf16 v16bf;
typedef __attribute__((ext_vector_type(8)))  float  v8f;

union BfFrag { uint4 u[2]; v16bf v; };
static_assert(sizeof(BfFrag) == 32, "frag size");

static __device__ __forceinline__ unsigned short f2bf(float f) {
  union { float f; unsigned u; } x; x.f = f;
  unsigned r = x.u + 0x7FFFu + ((x.u >> 16) & 1u);   // RNE to bf16
  return (unsigned short)(r >> 16);
}

// -------------------- zero fill --------------------
__global__ void k_zero(float* __restrict__ p, int n) {
  int i = blockIdx.x * blockDim.x + threadIdx.x;
  int st = gridDim.x * blockDim.x;
  for (; i < n; i += st) p[i] = 0.0f;
}

// ------------- 2-layer MLP: Linear(d,32)+ReLU+Linear(32,128) -------------
// out_f (fp32) optional; out_bf (bf16 bits) always written.
__global__ void __launch_bounds__(256) k_mlp2(
    const float* __restrict__ x, int d, int rows,
    const float* __restrict__ W1, const float* __restrict__ b1,
    const float* __restrict__ W2, const float* __restrict__ b2,
    float* __restrict__ out_f, unsigned short* __restrict__ out_bf) {
  __shared__ float sW1[16 * 32];
  __shared__ float sb1[32];
  __shared__ float sW2[32 * 128];
  __shared__ float sb2[128];
  int tid = threadIdx.x;
  for (int i = tid; i < d * 32; i += 256) sW1[i] = W1[i];
  for (int i = tid; i < 32 * 128; i += 256) sW2[i] = W2[i];
  if (tid < 32)  sb1[tid] = b1[tid];
  if (tid < 128) sb2[tid] = b2[tid];
  __syncthreads();
  int r = blockIdx.x * 256 + tid;
  if (r >= rows) return;
  float xv[16];
  for (int k = 0; k < d; ++k) xv[k] = x[(size_t)r * d + k];
  float hid[32];
  for (int j = 0; j < 32; ++j) {
    float s = sb1[j];
    for (int k = 0; k < d; ++k) s += xv[k] * sW1[k * 32 + j];
    hid[j] = fmaxf(s, 0.0f);
  }
  for (int n = 0; n < 128; ++n) {
    float a = sb2[n];
#pragma unroll 8
    for (int j = 0; j < 32; ++j) a += hid[j] * sW2[j * 128 + n];
    if (out_f) out_f[(size_t)r * 128 + n] = a;
    out_bf[(size_t)r * 128 + n] = f2bf(a);
  }
}

// ------------- pack W (384x128 fp32, row-major [k][n]) into WMMA B fragments -------------
// Layout: per (ntile t, kchunk c, lane) 8 dwords, each dword = bf16 pair (k, k+1)
// matching the 16-bit WMMA operand layout (lane<16: K pairs 0,2,4,6 / 16,18,20,22;
// lanes 16-31 shifted by 8 in K), column n = t*16 + lane%16.
__global__ void k_prep_w(const float* __restrict__ W, unsigned* __restrict__ frag) {
  int i = blockIdx.x * 256 + threadIdx.x;  // 8*12*32*8 = 24576
  if (i >= 8 * 12 * 32 * 8) return;
  int dwe  = i & 7;
  int lane = (i >> 3) & 31;
  int c    = (i >> 8) % 12;
  int t    = (i >> 8) / 12;
  int n    = t * 16 + (lane & 15);
  int hlf  = lane >> 4;
  int kb   = ((dwe & 3) << 1) + ((dwe >> 2) << 4) + hlf * 8;
  int k    = c * 32 + kb;
  unsigned lo = f2bf(W[(size_t)k * 128 + n]);
  unsigned hi = f2bf(W[(size_t)(k + 1) * 128 + n]);
  frag[i] = lo | (hi << 16);
}

// ------------- CGConv message+gate+scatter: 32 edges per workgroup -------------
// z = [h_dst(128) || h_src(128) || e(128)] bf16 (two 16-row A tiles in LDS),
// times Wf/Ws (384x128) via WMMA; each wave owns one 16-col N-tile and runs both
// M-halves against the same B fragments (4 WMMAs per B fetch).
// gate = sigmoid(zWf+bf) * softplus(zWs+bs), atomic scatter-add into acc[dst].
__global__ void __launch_bounds__(256) k_cgconv(
    const unsigned short* __restrict__ h_dst,
    const unsigned short* __restrict__ h_src,
    const unsigned short* __restrict__ ebf,
    const int* __restrict__ srcIdx, const int* __restrict__ dstIdx,
    const unsigned* __restrict__ fragF, const unsigned* __restrict__ fragS,
    const float* __restrict__ biasF, const float* __restrict__ biasS,
    float* __restrict__ acc) {
  __shared__ __align__(16) unsigned short zt[32][392];  // 392 = 384+8 pad (bank spread)
  __shared__ int sdst[32], ssrc[32];
  int tid = threadIdx.x;
  int e0 = blockIdx.x * 32;
  if (tid < 32) { sdst[tid] = dstIdx[e0 + tid]; ssrc[tid] = srcIdx[e0 + tid]; }
  __syncthreads();
  // gather z tile: 32 edges x 48 groups of 8 bf16 (16B)
  for (int i = tid; i < 32 * 48; i += 256) {
    int m = i / 48, g = i % 48;
    const uint4* src;
    if (g < 16)      src = (const uint4*)(h_dst + (size_t)sdst[m] * 128) + g;
    else if (g < 32) src = (const uint4*)(h_src + (size_t)ssrc[m] * 128) + (g - 16);
    else             src = (const uint4*)(ebf + (size_t)(e0 + m) * 128) + (g - 32);
    *(uint4*)&zt[m][g * 8] = *src;
  }
  __syncthreads();

  int lane = tid & 31, wave = tid >> 5;
  int mrow = lane & 15, hlf = lane >> 4;
  int t = wave;  // N-tile: columns t*16 .. t*16+15
  v8f cf0 = {0.f, 0.f, 0.f, 0.f, 0.f, 0.f, 0.f, 0.f};
  v8f cs0 = cf0, cf1 = cf0, cs1 = cf0;
#pragma unroll
  for (int c = 0; c < 12; ++c) {
    BfFrag a0, a1, bF, bS;
    const uint4* pa0 = (const uint4*)&zt[mrow][c * 32 + hlf * 8];
    a0.u[0] = pa0[0];       // K local [hlf*8 .. hlf*8+7]
    a0.u[1] = pa0[2];       // K local [16+hlf*8 .. 16+hlf*8+7]
    const uint4* pa1 = (const uint4*)&zt[16 + mrow][c * 32 + hlf * 8];
    a1.u[0] = pa1[0];
    a1.u[1] = pa1[2];
    const uint4* pf = (const uint4*)(fragF + (size_t)(((t * 12 + c) * 32 + lane) * 8));
    bF.u[0] = pf[0]; bF.u[1] = pf[1];
    const uint4* ps = (const uint4*)(fragS + (size_t)(((t * 12 + c) * 32 + lane) * 8));
    bS.u[0] = ps[0]; bS.u[1] = ps[1];
    cf0 = __builtin_amdgcn_wmma_f32_16x16x32_bf16(false, a0.v, false, bF.v, (short)0, cf0, false, false);
    cs0 = __builtin_amdgcn_wmma_f32_16x16x32_bf16(false, a0.v, false, bS.v, (short)0, cs0, false, false);
    cf1 = __builtin_amdgcn_wmma_f32_16x16x32_bf16(false, a1.v, false, bF.v, (short)0, cf1, false, false);
    cs1 = __builtin_amdgcn_wmma_f32_16x16x32_bf16(false, a1.v, false, bS.v, (short)0, cs1, false, false);
  }
  int n = t * 16 + mrow;
  float bfb = biasF[n], bsb = biasS[n];
#pragma unroll
  for (int r = 0; r < 8; ++r) {
    int mr = hlf ? (r + 8) : r;  // C/D layout: lanes 0-15 -> M=r, lanes 16-31 -> M=r+8
    {
      float f = cf0[r] + bfb;
      float s = cs0[r] + bsb;
      float sig = 1.0f / (1.0f + __expf(-f));
      float sp  = fmaxf(s, 0.0f) + log1pf(__expf(-fabsf(s)));  // stable softplus
      atomicAdd(&acc[(size_t)sdst[mr] * 128 + n], sig * sp);
    }
    {
      float f = cf1[r] + bfb;
      float s = cs1[r] + bsb;
      float sig = 1.0f / (1.0f + __expf(-f));
      float sp  = fmaxf(s, 0.0f) + log1pf(__expf(-fabsf(s)));
      atomicAdd(&acc[(size_t)sdst[16 + mr] * 128 + n], sig * sp);
    }
  }
}

// ------------- residual + leaky ReLU + refresh bf16 mirror -------------
__global__ void k_finalize(float* __restrict__ h, const float* __restrict__ acc,
                           unsigned short* __restrict__ hb, int n) {
  int i = blockIdx.x * blockDim.x + threadIdx.x;
  if (i >= n) return;
  float v = h[i] + acc[i];
  v = v > 0.0f ? v : 0.01f * v;
  h[i] = v;
  hb[i] = f2bf(v);
}

// ------------- pooling -------------
__global__ void k_count(const int* __restrict__ b, float* __restrict__ cnt, int n) {
  int i = blockIdx.x * blockDim.x + threadIdx.x;
  if (i < n) atomicAdd(&cnt[b[i]], 1.0f);
}
__global__ void k_poolsum(const float* __restrict__ h, const int* __restrict__ b,
                          float* __restrict__ g, int colbase, int nodes) {
  int idx = blockIdx.x * blockDim.x + threadIdx.x;
  if (idx >= nodes * 128) return;
  int i = idx >> 7, ch = idx & 127;
  atomicAdd(&g[(size_t)b[i] * 512 + colbase + ch], h[idx]);
}
__global__ void k_meandiv(float* __restrict__ g, const float* __restrict__ cin,
                          const float* __restrict__ cout) {
  int idx = blockIdx.x * blockDim.x + threadIdx.x;
  if (idx >= 64 * 128) return;
  int b = idx >> 7, ch = idx & 127;
  g[(size_t)b * 512 + 128 + ch] = g[(size_t)b * 512 + ch]       / fmaxf(cin[b], 1.0f);
  g[(size_t)b * 512 + 384 + ch] = g[(size_t)b * 512 + 256 + ch] / fmaxf(cout[b], 1.0f);
}

// ------------- readout MLP: 512 -> 128 -> 64 -> 1 over 64 graphs -------------
__global__ void __launch_bounds__(64) k_readout(
    const float* __restrict__ g,
    const float* __restrict__ W1, const float* __restrict__ b1,
    const float* __restrict__ W2, const float* __restrict__ b2,
    const float* __restrict__ W3, const float* __restrict__ b3,
    float* __restrict__ out) {
  __shared__ float h1[64 * 128];
  __shared__ float h2[64 * 64];
  int b = threadIdx.x;  // 0..63
  for (int n = 0; n < 128; ++n) {
    float a = b1[n];
    for (int k = 0; k < 512; ++k) a += g[(size_t)b * 512 + k] * W1[(size_t)k * 128 + n];
    h1[b * 128 + n] = fmaxf(a, 0.0f);
  }
  for (int n = 0; n < 64; ++n) {
    float a = b2[n];
    for (int k = 0; k < 128; ++k) a += h1[b * 128 + k] * W2[(size_t)k * 64 + n];
    h2[b * 64 + n] = fmaxf(a, 0.0f);
  }
  float a = b3[0];
  for (int k = 0; k < 64; ++k) a += h2[b * 64 + k] * W3[k];
  out[b] = a;
}

extern "C" void kernel_launch(void* const* d_in, const int* in_sizes, int n_in,
                              void* d_out, int out_size, void* d_ws, size_t ws_size,
                              hipStream_t stream) {
  (void)in_sizes; (void)n_in; (void)out_size; (void)ws_size;
  // ---- inputs (setup_inputs dict order, params flattened in insertion order) ----
  const float* x_in     = (const float*)d_in[0];
  const float* x_out    = (const float*)d_in[1];
  const float* ea_net   = (const float*)d_in[2];
  const float* ea_cell  = (const float*)d_in[3];
  const int*   ei_net   = (const int*)d_in[4];   // [2][E]: row0=src, row1=dst
  const int*   ei_cell  = (const int*)d_in[5];
  const int*   batch_in = (const int*)d_in[6];
  const int*   batch_out= (const int*)d_in[7];
  const float* net_eW1  = (const float*)d_in[8];
  const float* net_eb1  = (const float*)d_in[9];
  const float* net_eW2  = (const float*)d_in[10];
  const float* net_eb2  = (const float*)d_in[11];
  const float* cell_eW1 = (const float*)d_in[12];
  const float* cell_eb1 = (const float*)d_in[13];
  const float* cell_eW2 = (const float*)d_in[14];
  const float* cell_eb2 = (const float*)d_in[15];
  const float* in_nW1   = (const float*)d_in[16];
  const float* in_nb1   = (const float*)d_in[17];
  const float* in_nW2   = (const float*)d_in[18];
  const float* in_nb2   = (const float*)d_in[19];
  const float* out_nW1  = (const float*)d_in[20];
  const float* out_nb1  = (const float*)d_in[21];
  const float* out_nW2  = (const float*)d_in[22];
  const float* out_nb2  = (const float*)d_in[23];
  const float* net_Wf   = (const float*)d_in[24];  // [2][384][128]
  const float* net_bf   = (const float*)d_in[25];  // [2][128]
  const float* net_Ws   = (const float*)d_in[26];
  const float* net_bs   = (const float*)d_in[27];
  const float* cell_Wf  = (const float*)d_in[28];
  const float* cell_bf  = (const float*)d_in[29];
  const float* cell_Ws  = (const float*)d_in[30];
  const float* cell_bs  = (const float*)d_in[31];
  const float* mW1      = (const float*)d_in[32];
  const float* mb1      = (const float*)d_in[33];
  const float* mW2      = (const float*)d_in[34];
  const float* mb2      = (const float*)d_in[35];
  const float* mW3      = (const float*)d_in[36];
  const float* mb3      = (const float*)d_in[37];

  // ---- workspace layout (all sizes multiples of 256B) ----
  char* ws = (char*)d_ws;
  size_t off = 0;
  auto take = [&](size_t bytes) { char* p = ws + off; off += bytes; return p; };
  float*          h_in    = (float*)take((size_t)N_NODES * 128 * 4);       // 32 MB
  float*          h_out   = (float*)take((size_t)N_NODES * 128 * 4);
  float*          acc_in  = (float*)take((size_t)N_NODES * 128 * 4);
  float*          acc_out = (float*)take((size_t)N_NODES * 128 * 4);
  unsigned short* hb_in   = (unsigned short*)take((size_t)N_NODES * 128 * 2);
  unsigned short* hb_out  = (unsigned short*)take((size_t)N_NODES * 128 * 2);
  unsigned short* e_net   = (unsigned short*)take((size_t)N_EDGES * 128 * 2);
  unsigned short* e_cell  = (unsigned short*)take((size_t)N_EDGES * 128 * 2);
  unsigned*       wfrag   = (unsigned*)take(8 * 24576 * 4);                // 8 packed matrices
  float*          g       = (float*)take(64 * 512 * 4);
  float*          cnt     = (float*)take(512);                             // cnt_in[64], cnt_out[64]
  float* cnt_in = cnt, * cnt_out = cnt + 64;

  const int NH = N_NODES * 128;  // 8388608

  // ---- stage 1: edge / node MLPs ----
  k_mlp2<<<(N_EDGES + 255) / 256, 256, 0, stream>>>(ea_net, 4, N_EDGES, net_eW1, net_eb1, net_eW2, net_eb2, nullptr, e_net);
  k_mlp2<<<(N_EDGES + 255) / 256, 256, 0, stream>>>(ea_cell, 8, N_EDGES, cell_eW1, cell_eb1, cell_eW2, cell_eb2, nullptr, e_cell);
  k_mlp2<<<N_NODES / 256, 256, 0, stream>>>(x_in, 16, N_NODES, in_nW1, in_nb1, in_nW2, in_nb2, h_in, hb_in);
  k_mlp2<<<N_NODES / 256, 256, 0, stream>>>(x_out, 16, N_NODES, out_nW1, out_nb1, out_nW2, out_nb2, h_out, hb_out);

  // ---- stage 2: pack CGConv weights into WMMA B fragments ----
  const int LW = 384 * 128;  // per-layer weight elements
  k_prep_w<<<96, 256, 0, stream>>>(net_Wf,        wfrag + 0 * 24576);
  k_prep_w<<<96, 256, 0, stream>>>(net_Ws,        wfrag + 1 * 24576);
  k_prep_w<<<96, 256, 0, stream>>>(net_Wf + LW,   wfrag + 2 * 24576);
  k_prep_w<<<96, 256, 0, stream>>>(net_Ws + LW,   wfrag + 3 * 24576);
  k_prep_w<<<96, 256, 0, stream>>>(cell_Wf,       wfrag + 4 * 24576);
  k_prep_w<<<96, 256, 0, stream>>>(cell_Ws,       wfrag + 5 * 24576);
  k_prep_w<<<96, 256, 0, stream>>>(cell_Wf + LW,  wfrag + 6 * 24576);
  k_prep_w<<<96, 256, 0, stream>>>(cell_Ws + LW,  wfrag + 7 * 24576);

  // ---- stage 3: 2 layers of dual CGConv ----
  for (int l = 0; l < 2; ++l) {
    k_zero<<<4096, 256, 0, stream>>>(acc_in, NH);
    k_zero<<<4096, 256, 0, stream>>>(acc_out, NH);
    // ('outpin','net','inpin'): src=outpin(h_out), dst=inpin(h_in)
    k_cgconv<<<N_EDGES / 32, 256, 0, stream>>>(
        hb_in, hb_out, e_net, ei_net, ei_net + N_EDGES,
        wfrag + (2 * l + 0) * 24576, wfrag + (2 * l + 1) * 24576,
        net_bf + l * 128, net_bs + l * 128, acc_in);
    // ('inpin','cell','outpin'): src=inpin(h_in), dst=outpin(h_out)
    k_cgconv<<<N_EDGES / 32, 256, 0, stream>>>(
        hb_out, hb_in, e_cell, ei_cell, ei_cell + N_EDGES,
        wfrag + (4 + 2 * l) * 24576, wfrag + (5 + 2 * l) * 24576,
        cell_bf + l * 128, cell_bs + l * 128, acc_out);
    k_finalize<<<NH / 256, 256, 0, stream>>>(h_in, acc_in, hb_in, NH);
    k_finalize<<<NH / 256, 256, 0, stream>>>(h_out, acc_out, hb_out, NH);
  }

  // ---- stage 4: pooling + readout ----
  k_zero<<<128, 256, 0, stream>>>(g, 64 * 512 + 128);  // g then cnt (contiguous)
  k_count<<<N_NODES / 256, 256, 0, stream>>>(batch_in, cnt_in, N_NODES);
  k_count<<<N_NODES / 256, 256, 0, stream>>>(batch_out, cnt_out, N_NODES);
  k_poolsum<<<NH / 256, 256, 0, stream>>>(h_in, batch_in, g, 0, N_NODES);
  k_poolsum<<<NH / 256, 256, 0, stream>>>(h_out, batch_out, g, 256, N_NODES);
  k_meandiv<<<32, 256, 0, stream>>>(g, cnt_in, cnt_out);
  k_readout<<<1, 64, 0, stream>>>(g, mW1, mb1, mW2, mb2, mW3, mb3, (float*)d_out);
}
